// ContrastiveLearningModule_17248588660760
// MI455X (gfx1250) — matrix operand order repaired and checked by reading, not compiled, not run on hardware
//
#include <hip/hip_runtime.h>
#include <hip/hip_bf16.h>
#include <math.h>

typedef __attribute__((ext_vector_type(16))) _Float16 v16h;
typedef __attribute__((ext_vector_type(8)))  _Float16 v8h;
typedef __attribute__((ext_vector_type(8)))  float    v8f;

#define TT 4
#define VV 2
#define LL 3
#define NN 4000
#define DD 256
#define MM (TT * VV * LL * NN)      // 96000 rows
#define S_PRIO 100
#define S_REST 20
#define K_PRIO 50
#define K_REST 10
#define TOTAL_ANCHORS (2 * VV * LL * S_PRIO + 2 * VV * LL * S_REST)  // 1440
#define INV_TEMP 2.0f

// ---------------------------------------------------------------------------
// Kernel 1: W (f32, row-major [n][k]) -> Wt (f16, [k][n]) so B-fragments are
// contiguous across lanes (lane = N column, 16 K rows per lane).
// ---------------------------------------------------------------------------
__global__ __launch_bounds__(256) void prep_weights(
    const float* __restrict__ W1, const float* __restrict__ W2,
    _Float16* __restrict__ W1T, _Float16* __restrict__ W2T) {
  int e = blockIdx.x * 256 + threadIdx.x;          // 0 .. 2*65536-1
  int el = e & 65535;
  int k = el >> 8;
  int n = el & 255;
  if (e < 65536) {
    W1T[el] = (_Float16)W1[n * DD + k];
  } else {
    W2T[el] = (_Float16)W2[n * DD + k];
  }
}

// ---------------------------------------------------------------------------
// Kernel 2: fused 2-layer MLP, 16 rows per block, 4 waves, WMMA f16->f32.
//   x staged via GLOBAL_LOAD_ASYNC_TO_LDS_B128 (ASYNCcnt), then
//   h = relu(x @ W1^T + b1) ; z = h @ W2^T + b2 ; z -> P (unnormalized).
//   Loop is k-outer / tile-inner: one A fragment feeds 4 WMMAs.
// ---------------------------------------------------------------------------
__global__ __launch_bounds__(128) void proj_gemm(
    const float* __restrict__ E,
    const _Float16* __restrict__ W1T, const float* __restrict__ b1,
    const _Float16* __restrict__ W2T, const float* __restrict__ b2,
    float* __restrict__ P) {
  __shared__ float    xf32[16 * DD];   // 16 KB: raw f32 input tile (async)
  __shared__ _Float16 hbuf[16 * DD];   //  8 KB: f16 hidden tile

  const int tid  = threadIdx.x;
  const int lane = tid & 31;
  const int wave = tid >> 5;
  const int l15  = lane & 15;
  const int hsel = lane >> 4;              // 0 for lanes 0-15, 1 for 16-31
  const long rowBase = (long)blockIdx.x * 16;

  // --- async stage: 16 rows x 256 f32 (16 KB) -> LDS, no VGPR round-trip --
  {
    const unsigned long long gbase =
        (unsigned long long)(const void*)(E + rowBase * DD);
    // low 32 bits of the generic pointer == LDS byte offset (aperture rule)
    const unsigned ldsbase = (unsigned)(uintptr_t)(&xf32[0]);
#pragma unroll
    for (int i = 0; i < 8; ++i) {
      const int c = tid + i * 128;                     // 16B chunk id, 0..1023
      unsigned long long ga = gbase + (unsigned long long)c * 16u;
      unsigned la = ldsbase + (unsigned)c * 16u;
      asm volatile("global_load_async_to_lds_b128 %0, %1, off"
                   :: "v"(la), "v"(ga) : "memory");
    }
    asm volatile("s_wait_asynccnt 0x0" ::: "memory");
  }
  __syncthreads();

  // A-fragment (16x32) from row-major 16x256 f32 LDS tile, cvt to f16.
  // Lanes 0-15:  M=lane, K = k0+0..7 and k0+16..23
  // Lanes 16-31: M=lane-16, K = k0+8..15 and k0+24..31
  auto loadAf32 = [&](const float* buf, int k0) -> v16h {
    const float* p = buf + l15 * DD + k0 + hsel * 8;
    float4 f0 = ((const float4*)p)[0];
    float4 f1 = ((const float4*)p)[1];
    float4 f2 = ((const float4*)(p + 16))[0];
    float4 f3 = ((const float4*)(p + 16))[1];
    v16h a;
    a[0]  = (_Float16)f0.x; a[1]  = (_Float16)f0.y;
    a[2]  = (_Float16)f0.z; a[3]  = (_Float16)f0.w;
    a[4]  = (_Float16)f1.x; a[5]  = (_Float16)f1.y;
    a[6]  = (_Float16)f1.z; a[7]  = (_Float16)f1.w;
    a[8]  = (_Float16)f2.x; a[9]  = (_Float16)f2.y;
    a[10] = (_Float16)f2.z; a[11] = (_Float16)f2.w;
    a[12] = (_Float16)f3.x; a[13] = (_Float16)f3.y;
    a[14] = (_Float16)f3.z; a[15] = (_Float16)f3.w;
    return a;
  };
  // A-fragment from a row-major 16x256 f16 LDS tile (two b128 loads).
  auto loadAf16 = [&](const _Float16* buf, int k0) -> v16h {
    const _Float16* p = buf + l15 * DD + k0 + hsel * 8;
    v8h lo = *(const v8h*)(p);
    v8h hi = *(const v8h*)(p + 16);
    return __builtin_shufflevector(lo, hi, 0, 1, 2, 3, 4, 5, 6, 7,
                                   8, 9, 10, 11, 12, 13, 14, 15);
  };
  // B-fragment (32x16 f16): N = n0+lane%16; K = k0+8*hsel+0..7 and k0+16+8*hsel+0..7.
  auto loadB = [&](const _Float16* Wt, int k0, int n0) -> v16h {
    const _Float16* p = Wt + (k0 + hsel * 8) * DD + n0 + l15;
    v16h b;
#pragma unroll
    for (int s = 0; s < 8; ++s) {
      b[s] = p[s * DD];
      b[8 + s] = p[(16 + s) * DD];
    }
    return b;
  };

  // --- stage 1: h = relu(x @ W1^T + b1); wave owns cols [wave*64, +64) ----
  {
    v8f acc[4];
#pragma unroll
    for (int tI = 0; tI < 4; ++tI) {
      const float bv = b1[wave * 64 + tI * 16 + l15];
#pragma unroll
      for (int r = 0; r < 8; ++r) acc[tI][r] = bv;
    }
#pragma unroll
    for (int k0 = 0; k0 < DD; k0 += 32) {
      v16h a = loadAf32(xf32, k0);            // one A frag feeds 4 WMMAs
#pragma unroll
      for (int tI = 0; tI < 4; ++tI) {
        v16h b = loadB(W1T, k0, wave * 64 + tI * 16);
        acc[tI] = __builtin_amdgcn_wmma_f32_16x16x32_f16(
            false, a, false, b, (short)0, acc[tI], false, false);
      }
    }
    // relu + pack to f16 hidden (C layout: VGPR r -> row r + 8*hsel)
#pragma unroll
    for (int tI = 0; tI < 4; ++tI) {
      const int n0 = wave * 64 + tI * 16;
#pragma unroll
      for (int r = 0; r < 8; ++r) {
        float v = acc[tI][r] > 0.0f ? acc[tI][r] : 0.0f;
        hbuf[(r + 8 * hsel) * DD + n0 + l15] = (_Float16)v;
      }
    }
  }
  __syncthreads();

  // --- stage 2: z = h @ W2^T + b2, streamed to P (f32, unnormalized) ------
  {
    v8f acc[4];
#pragma unroll
    for (int tI = 0; tI < 4; ++tI) {
      const float bv = b2[wave * 64 + tI * 16 + l15];
#pragma unroll
      for (int r = 0; r < 8; ++r) acc[tI][r] = bv;
    }
#pragma unroll
    for (int k0 = 0; k0 < DD; k0 += 32) {
      v16h a = loadAf16(hbuf, k0);
#pragma unroll
      for (int tI = 0; tI < 4; ++tI) {
        v16h b = loadB(W2T, k0, wave * 64 + tI * 16);
        acc[tI] = __builtin_amdgcn_wmma_f32_16x16x32_f16(
            false, a, false, b, (short)0, acc[tI], false, false);
      }
    }
    float* dstBase = P + rowBase * DD;
#pragma unroll
    for (int tI = 0; tI < 4; ++tI) {
      const int n0 = wave * 64 + tI * 16;
#pragma unroll
      for (int r = 0; r < 8; ++r)
        dstBase[(r + 8 * hsel) * DD + n0 + l15] = acc[tI][r];
    }
  }
}

// ---------------------------------------------------------------------------
// Kernel 3: row-wise L2 normalize of P. One wave per row (8 rows/block).
// ---------------------------------------------------------------------------
__global__ __launch_bounds__(256) void norm_rows(float* __restrict__ P) {
  const long row = (long)blockIdx.x * 8 + (threadIdx.x >> 5);
  const int lane = threadIdx.x & 31;
  float* p = P + row * DD + lane * 8;
  float4 a = ((float4*)p)[0];
  float4 b = ((float4*)p)[1];
  float ss = a.x * a.x + a.y * a.y + a.z * a.z + a.w * a.w +
             b.x * b.x + b.y * b.y + b.z * b.z + b.w * b.w;
#pragma unroll
  for (int off = 16; off > 0; off >>= 1) ss += __shfl_xor(ss, off, 32);
  float inv = 1.0f / fmaxf(sqrtf(ss), 1e-12f);
  a.x *= inv; a.y *= inv; a.z *= inv; a.w *= inv;
  b.x *= inv; b.y *= inv; b.z *= inv; b.w *= inv;
  ((float4*)p)[0] = a;
  ((float4*)p)[1] = b;
}

// ---------------------------------------------------------------------------
// Kernel 4a: zero the scalar output (d_out is poisoned before timing).
// ---------------------------------------------------------------------------
__global__ void zero_out(float* out) { out[0] = 0.0f; }

// ---------------------------------------------------------------------------
// Kernel 4b: contrastive loss. One block per anchor (1440 blocks, 8 waves).
// ---------------------------------------------------------------------------
__global__ __launch_bounds__(256) void loss_kernel(
    const float* __restrict__ P,
    const int* __restrict__ idx_prio, const int* __restrict__ idx_rest,
    const int* __restrict__ neg_prio, const int* __restrict__ neg_rest,
    float* __restrict__ out) {
  __shared__ float zsh[DD];
  __shared__ float posSh[8];
  __shared__ float negSh[8];

  const int tid = threadIdx.x;
  const int lane = tid & 31;
  const int wave = tid >> 5;

  int bid = blockIdx.x;
  int t, v, l, s, S, K;
  const int* idx;
  const int* nidx;
  if (bid < 2 * VV * LL * S_PRIO) {                       // prio types 0,1
    t = bid / (VV * LL * S_PRIO);
    int rem = bid % (VV * LL * S_PRIO);
    S = S_PRIO; K = K_PRIO;
    v = rem / (LL * S); l = (rem / S) % LL; s = rem % S;
    idx  = idx_prio + ((t * VV + v) * LL + l) * S_PRIO;
    nidx = neg_prio + (((t * VV + v) * LL + l) * 3) * K_PRIO;
  } else {                                                // rest types 2,3
    int rb = bid - 2 * VV * LL * S_PRIO;
    int t2 = rb / (VV * LL * S_REST);
    int rem = rb % (VV * LL * S_REST);
    t = 2 + t2;
    S = S_REST; K = K_REST;
    v = rem / (LL * S); l = (rem / S) % LL; s = rem % S;
    idx  = idx_rest + ((t2 * VV + v) * LL + l) * S_REST;
    nidx = neg_rest + (((t2 * VV + v) * LL + l) * 3) * K_REST;
  }

  const int a = idx[s];
  const float* zptr = P + ((((long)t * VV + v) * LL + l) * NN + a) * DD;
  zsh[tid] = zptr[tid];
  __syncthreads();

  const float* zl = zsh + lane * 8;
  float z0 = zl[0], z1 = zl[1], z2 = zl[2], z3 = zl[3];
  float z4 = zl[4], z5 = zl[5], z6 = zl[6], z7 = zl[7];

  const int C = 6 + S + 3 * K;
  float posA = 0.0f, negA = 0.0f;
  for (int c = wave; c < C; c += 8) {
    const float* w;
    bool excl = false, isPos = false;
    if (c < 6) {                                   // cross-(view,layer) positives
      int x = c / LL, y = c % LL;
      excl = (x == v) && (y == l);
      w = P + ((((long)t * VV + x) * LL + y) * NN + a) * DD;
      isPos = true;
    } else if (c < 6 + S) {                        // in-type negatives
      int ttn = c - 6;
      excl = (ttn == s);
      int node = idx[ttn];
      w = P + ((((long)t * VV + v) * LL + l) * NN + node) * DD;
    } else {                                       // cross-type negatives
      int j = c - 6 - S;
      int o = j / K, k = j % K;
      int ot = o + (o >= t ? 1 : 0);               // others = types != t, sorted
      int node = nidx[o * K + k];
      w = P + ((((long)ot * VV + v) * LL + l) * NN + node) * DD;
    }
    const float4* wv = (const float4*)(w + lane * 8);
    float4 w0 = wv[0], w1 = wv[1];
    float dp = z0 * w0.x + z1 * w0.y + z2 * w0.z + z3 * w0.w +
               z4 * w1.x + z5 * w1.y + z6 * w1.z + z7 * w1.w;
#pragma unroll
    for (int off = 16; off > 0; off >>= 1) dp += __shfl_xor(dp, off, 32);
    float e = excl ? 0.0f : __expf(dp * INV_TEMP);
    if (isPos) posA += e; else negA += e;
  }
  if (lane == 0) { posSh[wave] = posA; negSh[wave] = negA; }
  __syncthreads();
  if (tid == 0) {
    float pos = 0.0f, neg = 0.0f;
#pragma unroll
    for (int i = 0; i < 8; ++i) { pos += posSh[i]; neg += negSh[i]; }
    float loss = -__logf(pos / (pos + neg));
    atomicAdd(out, loss * (1.0f / (float)TOTAL_ANCHORS));
  }
}

// ---------------------------------------------------------------------------
extern "C" void kernel_launch(void* const* d_in, const int* in_sizes, int n_in,
                              void* d_out, int out_size, void* d_ws, size_t ws_size,
                              hipStream_t stream) {
  const float* E  = (const float*)d_in[0];
  const float* W1 = (const float*)d_in[1];
  const float* b1 = (const float*)d_in[2];
  const float* W2 = (const float*)d_in[3];
  const float* b2 = (const float*)d_in[4];
  const int* idx_prio = (const int*)d_in[5];
  const int* idx_rest = (const int*)d_in[6];
  const int* neg_prio = (const int*)d_in[7];
  const int* neg_rest = (const int*)d_in[8];
  float* out = (float*)d_out;

  char* ws = (char*)d_ws;
  _Float16* W1T = (_Float16*)ws;                       // 128 KB
  _Float16* W2T = (_Float16*)(ws + 131072);            // 128 KB
  float* P = (float*)(ws + 262144);                    // 96000*256*4 ≈ 94 MB

  prep_weights<<<512, 256, 0, stream>>>(W1, W2, W1T, W2T);
  proj_gemm<<<MM / 16, 128, 0, stream>>>(E, W1T, b1, W2T, b2, P);
  norm_rows<<<MM / 8, 256, 0, stream>>>(P);
  zero_out<<<1, 1, 0, stream>>>(out);
  loss_kernel<<<TOTAL_ANCHORS, 256, 0, stream>>>(P, idx_prio, idx_rest,
                                                 neg_prio, neg_rest, out);
}
